// CustomScaledDotProductAttention_32143535243940
// MI455X (gfx1250) — matrix operand order repaired and checked by reading, not compiled, not run on hardware
//
#include <hip/hip_runtime.h>
#include <math.h>

typedef __attribute__((ext_vector_type(16))) _Float16 v16h;
typedef __attribute__((ext_vector_type(8)))  _Float16 v8h;
typedef __attribute__((ext_vector_type(4)))  _Float16 v4h;
typedef __attribute__((ext_vector_type(8)))  float    v8f;
typedef __attribute__((ext_vector_type(4)))  float    v4f;

#define EMBED 1024
#define NHEAD 16
#define HDIM  64
#define SEQ   2048
#define BATCH 2
#define MTOT  (BATCH*SEQ)   // 4096
#define QKVN  (3*EMBED)     // 3072
#define ATTN_SCALE 0.125f   // 64^-0.5

// ---------- WMMA fragment helpers (CDNA5 ISA 7.12.2 layouts, wave32) ----------

__device__ __forceinline__ v16h frag_cat(v8h lo, v8h hi) {
  return __builtin_shufflevector(lo, hi, 0,1,2,3,4,5,6,7,8,9,10,11,12,13,14,15);
}

// A operand, 16x32 f16. lane&15 = row M; lanes>=16 hold K base +8.
__device__ __forceinline__ v16h load_a(const _Float16* base, long ld, int lane) {
  const _Float16* p = base + (long)(lane & 15) * ld + ((lane >> 4) << 3);
  v8h lo = *(const v8h*)p;
  v8h hi = *(const v8h*)(p + 16);
  return frag_cat(lo, hi);
}

// B operand, 32x16 f16. lane&15 = col N; lanes>=16 hold K base +8 (same K order as load_a).
// element (k,n) lives at base[n*ld + k] (each lane reads two 8-wide K runs).
__device__ __forceinline__ v16h load_b(const _Float16* base, long ld, int lane) {
  const _Float16* p = base + (long)(lane & 15) * ld + ((lane >> 4) << 3);
  v8h lo = *(const v8h*)p;
  v8h hi = *(const v8h*)(p + 16);
  return frag_cat(lo, hi);
}

__device__ __forceinline__ v8f wmma_f16(v16h a, v16h b, v8f c) {
  return __builtin_amdgcn_wmma_f32_16x16x32_f16(false, a, false, b, (short)0, c, false, false);
}

// ---------- shared 32x64-per-wave GEMM core, K=EMBED, ping-pong pipelined ----------
// C[32x64] += A[32xK] * B[K x 64] with A row-major (ld EMBED) at Ab,
// B stored as weight rows (col n of B = contig K at Bb + n*EMBED).

__device__ __forceinline__ void gemm_core_32x64(
    const _Float16* __restrict__ Ab, const _Float16* __restrict__ Bb,
    int lane, v8f acc[2][4])
{
  v16h a0[2], b0[4], a1[2], b1[4];
  a0[0] = load_a(Ab,                  EMBED, lane);
  a0[1] = load_a(Ab + 16L * EMBED,    EMBED, lane);
  #pragma unroll
  for (int t = 0; t < 4; t++) b0[t] = load_b(Bb + (long)t * 16 * EMBED, EMBED, lane);

  for (int k0 = 0; k0 < EMBED - 64; k0 += 64) {
    // buf1 <- k0+32 (in flight while buf0 MMAs run)
    a1[0] = load_a(Ab + k0 + 32,               EMBED, lane);
    a1[1] = load_a(Ab + 16L * EMBED + k0 + 32, EMBED, lane);
    #pragma unroll
    for (int t = 0; t < 4; t++) b1[t] = load_b(Bb + (long)t * 16 * EMBED + k0 + 32, EMBED, lane);
    #pragma unroll
    for (int i = 0; i < 2; i++)
      #pragma unroll
      for (int t = 0; t < 4; t++) acc[i][t] = wmma_f16(a0[i], b0[t], acc[i][t]);
    // buf0 <- k0+64
    a0[0] = load_a(Ab + k0 + 64,               EMBED, lane);
    a0[1] = load_a(Ab + 16L * EMBED + k0 + 64, EMBED, lane);
    #pragma unroll
    for (int t = 0; t < 4; t++) b0[t] = load_b(Bb + (long)t * 16 * EMBED + k0 + 64, EMBED, lane);
    #pragma unroll
    for (int i = 0; i < 2; i++)
      #pragma unroll
      for (int t = 0; t < 4; t++) acc[i][t] = wmma_f16(a1[i], b1[t], acc[i][t]);
  }
  // tail: buf0 holds K-step EMBED-64; load EMBED-32 into buf1
  a1[0] = load_a(Ab + EMBED - 32,               EMBED, lane);
  a1[1] = load_a(Ab + 16L * EMBED + EMBED - 32, EMBED, lane);
  #pragma unroll
  for (int t = 0; t < 4; t++) b1[t] = load_b(Bb + (long)t * 16 * EMBED + EMBED - 32, EMBED, lane);
  #pragma unroll
  for (int i = 0; i < 2; i++)
    #pragma unroll
    for (int t = 0; t < 4; t++) acc[i][t] = wmma_f16(a0[i], b0[t], acc[i][t]);
  #pragma unroll
  for (int i = 0; i < 2; i++)
    #pragma unroll
    for (int t = 0; t < 4; t++) acc[i][t] = wmma_f16(a1[i], b1[t], acc[i][t]);
}

// ---------- f32 -> f16 convert (vectorized x4) ----------

__global__ void cvt_f16(const v4f* __restrict__ in, v4h* __restrict__ out, int n4) {
  int i = blockIdx.x * blockDim.x + threadIdx.x;
  if (i < n4) {
    v4f x = in[i];
    v4h y;
    y[0] = (_Float16)x[0]; y[1] = (_Float16)x[1];
    y[2] = (_Float16)x[2]; y[3] = (_Float16)x[3];
    out[i] = y;
  }
}

// ---------- QKV projection: qkv = x @ W^T + b, scattered to q/k/v^T ----------

__global__ __launch_bounds__(256) void qkv_gemm(
    const _Float16* __restrict__ X, const _Float16* __restrict__ W,
    const float* __restrict__ bias,
    _Float16* __restrict__ qh, _Float16* __restrict__ kh, _Float16* __restrict__ vth)
{
  const int lane = threadIdx.x & 31, wave = threadIdx.x >> 5;
  const int row0 = blockIdx.y * 128 + (wave >> 1) * 32;
  const int col0 = blockIdx.x * 128 + (wave & 1) * 64;
  v8f acc[2][4] = {};
  gemm_core_32x64(X + (long)row0 * EMBED, W + (long)col0 * EMBED, lane, acc);

  const int colL = lane & 15, rbase = (lane >> 4) << 3;
  #pragma unroll
  for (int t = 0; t < 4; t++) {
    const int col = col0 + t * 16 + colL;
    const float bv = bias[col];
    const int s = col >> 10;          // 0=q 1=k 2=v
    const int c = col & 1023;
    const int h = c >> 6, d = c & 63;
    #pragma unroll
    for (int i = 0; i < 2; i++) {
      #pragma unroll
      for (int r = 0; r < 8; r++) {
        const int row = row0 + i * 16 + rbase + r;
        const int b = row >> 11, npos = row & 2047;
        const float val = acc[i][t][r] + bv;
        const long bh = (long)(b * NHEAD + h);
        if (s == 0)      qh [(bh * SEQ + npos) * HDIM + d] = (_Float16)(val * ATTN_SCALE);
        else if (s == 1) kh [(bh * SEQ + npos) * HDIM + d] = (_Float16)val;
        else             vth[(bh * HDIM + d) * SEQ + npos] = (_Float16)val;
      }
    }
  }
}

// ---------- Flash attention: per (b,h), 16 query rows per wave ----------

__global__ __launch_bounds__(256) void attn_fa(
    const _Float16* __restrict__ qh, const _Float16* __restrict__ kh,
    const _Float16* __restrict__ vth, _Float16* __restrict__ attnh)
{
  __shared__ __align__(32) _Float16 psh[8][16 * 32];   // per-wave P staging (C->A transpose)
  const int lane = threadIdx.x & 31, wave = threadIdx.x >> 5;
  const int bh = blockIdx.y;                            // b*NHEAD + h
  const int q0 = blockIdx.x * 128 + wave * 16;
  const _Float16* qp = qh  + (long)bh * SEQ * HDIM;
  const _Float16* kp = kh  + (long)bh * SEQ * HDIM;
  const _Float16* vp = vth + (long)bh * HDIM * SEQ;

  const v16h qa0 = load_a(qp + (long)q0 * HDIM,      HDIM, lane);
  const v16h qa1 = load_a(qp + (long)q0 * HDIM + 32, HDIM, lane);

  float mrow[8], lrow[8];
  v8f o[4] = {};
  #pragma unroll
  for (int r = 0; r < 8; r++) { mrow[r] = -INFINITY; lrow[r] = 0.f; }

  _Float16* pw = &psh[wave][0];
  const int colL = lane & 15, rbase = (lane >> 4) << 3;

  auto loadK = [&](int j, v16h kb[4]) {
    kb[0] = load_b(kp + (long)j * HDIM,            HDIM, lane);
    kb[1] = load_b(kp + (long)j * HDIM + 32,       HDIM, lane);
    kb[2] = load_b(kp + (long)(j + 16) * HDIM,     HDIM, lane);
    kb[3] = load_b(kp + (long)(j + 16) * HDIM + 32, HDIM, lane);
  };
  auto loadV = [&](int j, v16h vb[4]) {
    #pragma unroll
    for (int t = 0; t < 4; t++) vb[t] = load_b(vp + (long)(t * 16) * SEQ + j, SEQ, lane);
  };

  // process one 32-key tile: scores via already-resident kb, online softmax,
  // P staged through per-wave LDS (same-wave DS ops are in-order; no barrier),
  // then P(16x32) @ V(32x64) with already-in-flight vb.
  auto process = [&](const v16h* kb, const v16h* vb) {
    v8f s0 = {}, s1 = {};
    s0 = wmma_f16(qa0, kb[0], s0);
    s0 = wmma_f16(qa1, kb[1], s0);
    s1 = wmma_f16(qa0, kb[2], s1);
    s1 = wmma_f16(qa1, kb[3], s1);
    #pragma unroll
    for (int r = 0; r < 8; r++) {
      float mx = fmaxf(s0[r], s1[r]);
      mx = fmaxf(mx, __shfl_xor(mx, 1, 32));
      mx = fmaxf(mx, __shfl_xor(mx, 2, 32));
      mx = fmaxf(mx, __shfl_xor(mx, 4, 32));
      mx = fmaxf(mx, __shfl_xor(mx, 8, 32));
      const float mnew  = fmaxf(mrow[r], mx);
      const float p0    = __expf(s0[r] - mnew);
      const float p1    = __expf(s1[r] - mnew);
      const float alpha = __expf(mrow[r] - mnew);
      float rs = p0 + p1;
      rs += __shfl_xor(rs, 1, 32);
      rs += __shfl_xor(rs, 2, 32);
      rs += __shfl_xor(rs, 4, 32);
      rs += __shfl_xor(rs, 8, 32);
      lrow[r] = lrow[r] * alpha + rs;
      mrow[r] = mnew;
      #pragma unroll
      for (int t = 0; t < 4; t++) o[t][r] *= alpha;
      s0[r] = p0; s1[r] = p1;
    }
    #pragma unroll
    for (int r = 0; r < 8; r++) {
      pw[(rbase + r) * 32 + colL]      = (_Float16)s0[r];
      pw[(rbase + r) * 32 + 16 + colL] = (_Float16)s1[r];
    }
    const v16h pa = load_a(pw, 32, lane);
    #pragma unroll
    for (int t = 0; t < 4; t++) o[t] = wmma_f16(pa, vb[t], o[t]);
  };

  v16h kbA[4], kbB[4], vb[4];
  loadK(0, kbA);
  for (int j0 = 0; j0 < SEQ - 64; j0 += 64) {
    // prefetch tile j0+64 (global_prefetch_b8): K rows + V rows
    __builtin_prefetch(kp + (long)(j0 + 64) * HDIM + lane * HDIM, 0, 1);
    __builtin_prefetch(vp + (long)lane        * SEQ + j0 + 64, 0, 1);
    __builtin_prefetch(vp + (long)(lane + 32) * SEQ + j0 + 64, 0, 1);

    loadV(j0, vb);
    loadK(j0 + 32, kbB);           // next tile's K in flight during this tile
    process(kbA, vb);
    loadV(j0 + 32, vb);
    loadK(j0 + 64, kbA);
    process(kbB, vb);
  }
  loadV(SEQ - 64, vb);
  loadK(SEQ - 32, kbB);
  process(kbA, vb);
  loadV(SEQ - 32, vb);
  process(kbB, vb);

  const int b = bh >> 4, h = bh & 15;
  float rinv[8];
  #pragma unroll
  for (int r = 0; r < 8; r++) rinv[r] = 1.0f / lrow[r];
  #pragma unroll
  for (int t = 0; t < 4; t++) {
    #pragma unroll
    for (int r = 0; r < 8; r++) {
      const int row = q0 + rbase + r;
      const int d   = t * 16 + colL;
      attnh[((long)(b * SEQ + row)) * EMBED + h * HDIM + d] = (_Float16)(o[t][r] * rinv[r]);
    }
  }
}

// ---------- Output projection: out = attn @ w_out^T + b_out (f32 out) ----------

__global__ __launch_bounds__(256) void out_gemm(
    const _Float16* __restrict__ A, const _Float16* __restrict__ W,
    const float* __restrict__ bias, float* __restrict__ out)
{
  const int lane = threadIdx.x & 31, wave = threadIdx.x >> 5;
  const int row0 = blockIdx.y * 128 + (wave >> 1) * 32;
  const int col0 = blockIdx.x * 128 + (wave & 1) * 64;
  v8f acc[2][4] = {};
  gemm_core_32x64(A + (long)row0 * EMBED, W + (long)col0 * EMBED, lane, acc);

  const int colL = lane & 15, rbase = (lane >> 4) << 3;
  #pragma unroll
  for (int t = 0; t < 4; t++) {
    const int col = col0 + t * 16 + colL;
    const float bv = bias[col];
    #pragma unroll
    for (int i = 0; i < 2; i++) {
      #pragma unroll
      for (int r = 0; r < 8; r++) {
        const int row = row0 + i * 16 + rbase + r;
        out[(long)row * EMBED + col] = acc[i][t][r] + bv;
      }
    }
  }
}

// ---------- host launch ----------

extern "C" void kernel_launch(void* const* d_in, const int* in_sizes, int n_in,
                              void* d_out, int out_size, void* d_ws, size_t ws_size,
                              hipStream_t stream) {
  (void)in_sizes; (void)n_in; (void)out_size; (void)ws_size;
  const float* x     = (const float*)d_in[0];
  const float* w_qkv = (const float*)d_in[1];
  const float* b_qkv = (const float*)d_in[2];
  const float* w_out = (const float*)d_in[3];
  const float* b_out = (const float*)d_in[4];

  char* ws = (char*)d_ws;                       // ~48 MB used
  _Float16* xh    = (_Float16*)(ws + 0);        //  8 MB  x     f16 [4096,1024]
  _Float16* wqkvh = (_Float16*)(ws + 8388608);  //  6 MB  w_qkv f16 [3072,1024]
  _Float16* wouth = (_Float16*)(ws + 14680064); //  2 MB  w_out f16 [1024,1024]
  _Float16* qh    = (_Float16*)(ws + 16777216); //  8 MB  q     f16 [B,H,N,D] (pre-scaled)
  _Float16* kh    = (_Float16*)(ws + 25165824); //  8 MB  k     f16 [B,H,N,D]
  _Float16* vth   = (_Float16*)(ws + 33554432); //  8 MB  v^T   f16 [B,H,D,N]
  _Float16* attnh = (_Float16*)(ws + 41943040); //  8 MB  attn  f16 [B,N,C]

  const int nx4  = MTOT * EMBED / 4;
  const int nwq4 = QKVN * EMBED / 4;
  const int nwo4 = EMBED * EMBED / 4;
  cvt_f16<<<(nx4  + 255) / 256, 256, 0, stream>>>((const v4f*)x,     (v4h*)xh,    nx4);
  cvt_f16<<<(nwq4 + 255) / 256, 256, 0, stream>>>((const v4f*)w_qkv, (v4h*)wqkvh, nwq4);
  cvt_f16<<<(nwo4 + 255) / 256, 256, 0, stream>>>((const v4f*)w_out, (v4h*)wouth, nwo4);

  qkv_gemm<<<dim3(QKVN / 128, MTOT / 128), 256, 0, stream>>>(xh, wqkvh, b_qkv, qh, kh, vth);
  attn_fa <<<dim3(SEQ / 128, BATCH * NHEAD), 256, 0, stream>>>(qh, kh, vth, attnh);
  out_gemm<<<dim3(EMBED / 128, MTOT / 128), 256, 0, stream>>>(attnh, wouth, b_out, (float*)d_out);
}
